// EdgeNetwork_55705725829532
// MI455X (gfx1250) — compile-verified, matrix operand order checked by reading, not run on hardware
//
#include <hip/hip_runtime.h>
#include <hip/hip_bf16.h>

#define D       128     // hidden dim d
#define EDIM    16      // edge feature dim
#define EH      128     // edge MLP hidden (K-rows of W2)
#define MT      64      // edges per block
#define THREADS 256     // 8 waves (wave32)
#define W2ROW   (D * D) // 16384 elements per W2 row
#define KROWS   (EH + 1)

typedef __attribute__((ext_vector_type(16))) _Float16 v16h;
typedef __attribute__((ext_vector_type(8)))  _Float16 v8h;
typedef __attribute__((ext_vector_type(4)))  _Float16 v4h;
typedef __attribute__((ext_vector_type(8)))  float    v8f;
typedef __attribute__((ext_vector_type(4)))  int      v4i;

typedef __attribute__((address_space(1))) v4i* gv4i_p;   // global int4*
typedef __attribute__((address_space(3))) v4i* lv4i_p;   // LDS int4* (32-bit)

// ---- CDNA5 async global->LDS copy (ASYNCcnt-tracked), 16B per lane ----
__device__ __forceinline__ void async_copy_b128(const _Float16* g, _Float16* l) {
#if __has_builtin(__builtin_amdgcn_global_load_async_to_lds_b128)
    // builtin signature (probe-learned): (int4 AS1*, int4 AS3*, imm offset, imm cpol)
    gv4i_p gp = (gv4i_p)(unsigned long long)(const void*)g;          // 64-bit global addr
    lv4i_p lp = (lv4i_p)(unsigned)(unsigned long long)(const void*)l; // low 32 bits = LDS offset
    __builtin_amdgcn_global_load_async_to_lds_b128(gp, lp, 0, 0);
#else
    unsigned loff = (unsigned)(unsigned long long)(const void*)l;
    asm volatile("global_load_async_to_lds_b128 %0, %1, off"
                 :: "v"(loff), "v"((unsigned long long)g) : "memory");
#endif
}
__device__ __forceinline__ void wait_async0() {
#if __has_builtin(__builtin_amdgcn_s_wait_asynccnt)
    __builtin_amdgcn_s_wait_asynccnt(0);
#else
    asm volatile("s_wait_asynccnt 0" ::: "memory");
#endif
}

// ---- Pre-pass: [W2 ; b2] f32 -> f16 workspace (129 x 16384), coalesced ----
__global__ void cvt_w2_f16(const float* __restrict__ W2, const float* __restrict__ b2,
                           _Float16* __restrict__ w2h) {
    const size_t n4w2 = (size_t)EH * W2ROW / 4;          // 524288 float4 of W2
    const size_t n4   = (size_t)KROWS * W2ROW / 4;       // + 4096 float4 of b2
    for (size_t f = blockIdx.x * (size_t)blockDim.x + threadIdx.x; f < n4;
         f += (size_t)gridDim.x * blockDim.x) {
        const float4 v = (f < n4w2) ? ((const float4*)W2)[f]
                                    : ((const float4*)b2)[f - n4w2];
        v4h hv; hv[0] = (_Float16)v.x; hv[1] = (_Float16)v.y;
                hv[2] = (_Float16)v.z; hv[3] = (_Float16)v.w;
        *(v4h*)(w2h + 4 * f) = hv;
    }
}

// Fused EdgeNetwork: messages = einsum('ek,kij,ej->ei', relu(ef@W1+b1)|1, [W2;b2], h_w)
// as one outer-product GEMM (M=E, N=128, K=129*128) on v_wmma_f32_16x16x32_f16.
// USE_ASYNC: stage f16 W2 rows with global_load_async_to_lds_b128 (double-buffered).
template<bool USE_ASYNC>
__global__ __launch_bounds__(THREADS)
void edge_net_wmma_kernel(const float* __restrict__ h_w,
                          const float* __restrict__ ef,
                          const float* __restrict__ W1,
                          const float* __restrict__ b1,
                          const float* __restrict__ W2,
                          const float* __restrict__ b2,
                          const _Float16* __restrict__ w2h,
                          float* __restrict__ out, int E)
{
    __shared__ __align__(16) _Float16 s_hidden[MT * (EH + 1)]; // [64][129] f16 (+1.0 bias ch)
    __shared__ __align__(16) _Float16 s_w2row[2][W2ROW];       // double-buffered f16 W2 rows

    const int tid   = threadIdx.x;
    const int wave  = tid >> 5;
    const int lane  = tid & 31;
    const int lanem = lane & 15;
    const int half  = lane >> 4;
    const int mw    = wave & 3;   // 4 M-groups of 16 edges
    const int nw    = wave >> 2;  // 2 N-groups of 64 output cols
    const int e0    = blockIdx.x * MT;

    // ---------- Stage 0: hidden = relu(ef @ W1 + b1), f16 into LDS ----------
    {
        float* sf = (float*)&s_w2row[0][0]; // reuse: [0,2048)=W1, [2048,2176)=b1, [2176,3200)=ef
        for (int idx = tid; idx < EDIM * EH; idx += THREADS) sf[idx] = W1[idx];
        if (tid < EH) sf[2048 + tid] = b1[tid];
        for (int idx = tid; idx < MT * EDIM; idx += THREADS) {
            int el = idx >> 4, t = idx & 15;
            int e = e0 + el; if (e >= E) e = E - 1;
            sf[2176 + idx] = ef[e * EDIM + t];
        }
        __syncthreads();
        #pragma unroll
        for (int r = 0; r < (MT * EH) / THREADS; ++r) {
            int oi = r * THREADS + tid;
            int el = oi >> 7, k = oi & 127;
            float acc = sf[2048 + k];
            #pragma unroll
            for (int t = 0; t < EDIM; ++t) acc += sf[2176 + el * EDIM + t] * sf[t * EH + k];
            acc = acc > 0.0f ? acc : 0.0f;
            s_hidden[el * (EH + 1) + k] = (_Float16)acc;
        }
        if (tid < MT) s_hidden[tid * (EH + 1) + EH] = (_Float16)1.0f; // bias channel
        __syncthreads();
    }

    // ---------- Per-lane h_w fragments in WMMA A-layout (loaded once) ----------
    v16h hwf[4];
    {
        int e = e0 + 16 * mw + lanem; if (e >= E) e = E - 1;
        const float* hp = h_w + (size_t)e * D;
        #pragma unroll
        for (int jc = 0; jc < 4; ++jc) {
            const int jb = 32 * jc + 8 * half;
            v16h f;
            #pragma unroll
            for (int h = 0; h < 8; ++h) f[h]     = (_Float16)hp[jb + h];
            #pragma unroll
            for (int h = 0; h < 8; ++h) f[8 + h] = (_Float16)hp[jb + 16 + h];
            hwf[jc] = f;
        }
    }

    v8f acc[4] = {};
    const int srow = 16 * mw + lanem;

    // ---------- Prologue: stage row 0 into buffer 0 ----------
    if constexpr (USE_ASYNC) {
        #pragma unroll
        for (int r = 0; r < W2ROW / (8 * THREADS); ++r) {    // 8 x b128 per thread
            int f = tid + r * THREADS;
            async_copy_b128(w2h + 8 * f, &s_w2row[0][8 * f]);
        }
        wait_async0();
    } else {
        const float4* src = (const float4*)W2;
        #pragma unroll
        for (int r = 0; r < W2ROW / (4 * THREADS); ++r) {
            int f = tid + r * THREADS;
            const float4 v = src[f];
            v4h hv; hv[0] = (_Float16)v.x; hv[1] = (_Float16)v.y;
                    hv[2] = (_Float16)v.z; hv[3] = (_Float16)v.w;
            *(v4h*)(&s_w2row[0][4 * f]) = hv;
        }
    }
    __syncthreads();

    // ---------- Main K loop: 128 W2 rows + 1 virtual b2 row, double-buffered ----------
    for (int k = 0; k < KROWS; ++k) {
        const int cur = k & 1, nxt = cur ^ 1;
        const bool more = (k + 1) < KROWS;

        float4 pre[16];                       // live only in !USE_ASYNC instantiation
        if constexpr (USE_ASYNC) {
            if (more) {                       // async DMA row k+1 -> buffer nxt
                const _Float16* src = w2h + (size_t)(k + 1) * W2ROW;
                #pragma unroll
                for (int r = 0; r < 8; ++r) {
                    int f = tid + r * THREADS;
                    async_copy_b128(src + 8 * f, &s_w2row[nxt][8 * f]);
                }
            }
        } else {
            if (more) {
                const float* srcf = (k + 1 < EH) ? (W2 + (size_t)(k + 1) * W2ROW) : b2;
                const float4* src = (const float4*)srcf;
                #pragma unroll
                for (int r = 0; r < 16; ++r) pre[r] = src[tid + r * THREADS];
            }
        }

        // compute on row k from LDS buffer `cur`
        const _Float16 sh = s_hidden[srow * (EH + 1) + k];
        #pragma unroll
        for (int jc = 0; jc < 4; ++jc) {
            v16h a = hwf[jc] * sh;            // U fragment = hidden[e,k]*h_w[e,j]
            #pragma unroll
            for (int nt = 0; nt < 4; ++nt) {
                const int ib = (64 * nw + 16 * nt + lanem) * D + 32 * jc + 8 * half;
                v8h lo = *(const v8h*)(&s_w2row[cur][ib]);
                v8h hi = *(const v8h*)(&s_w2row[cur][ib + 16]);
                v16h b = __builtin_shufflevector(lo, hi, 0,1,2,3,4,5,6,7,
                                                          8,9,10,11,12,13,14,15);
                acc[nt] = __builtin_amdgcn_wmma_f32_16x16x32_f16(
                    false, a, false, b, (short)0, acc[nt], false, false);
            }
        }

        if constexpr (USE_ASYNC) {
            wait_async0();                    // row k+1 landed in LDS
        } else {
            if (more) {
                #pragma unroll
                for (int r = 0; r < 16; ++r) {
                    int f = tid + r * THREADS;
                    v4h hv; hv[0] = (_Float16)pre[r].x; hv[1] = (_Float16)pre[r].y;
                            hv[2] = (_Float16)pre[r].z; hv[3] = (_Float16)pre[r].w;
                    *(v4h*)(&s_w2row[nxt][4 * f]) = hv;
                }
            }
        }
        __syncthreads();
    }

    // ---------- Epilogue: C layout -> messages (f32), tail-guarded ----------
    #pragma unroll
    for (int nt = 0; nt < 4; ++nt) {
        const int i = 64 * nw + 16 * nt + lanem;
        #pragma unroll
        for (int r = 0; r < 8; ++r) {
            const int e = e0 + 16 * mw + 8 * half + r;
            if (e < E) out[(size_t)e * D + i] = acc[nt][r];
        }
    }
}

extern "C" void kernel_launch(void* const* d_in, const int* in_sizes, int n_in,
                              void* d_out, int out_size, void* d_ws, size_t ws_size,
                              hipStream_t stream) {
    // inputs: h_v(unused), h_w, edge_features, W1, b1, W2, b2
    const float* h_w = (const float*)d_in[1];
    const float* ef  = (const float*)d_in[2];
    const float* W1  = (const float*)d_in[3];
    const float* b1  = (const float*)d_in[4];
    const float* W2  = (const float*)d_in[5];
    const float* b2  = (const float*)d_in[6];
    float* out = (float*)d_out;
    const int E = in_sizes[1] / D;
    const int blocks = (E + MT - 1) / MT;
    const size_t need = (size_t)KROWS * W2ROW * sizeof(_Float16);  // ~4.03 MiB

    if (ws_size >= need) {
        _Float16* w2h = (_Float16*)d_ws;
        cvt_w2_f16<<<512, 256, 0, stream>>>(W2, b2, w2h);
        edge_net_wmma_kernel<true><<<blocks, THREADS, 0, stream>>>(
            h_w, ef, W1, b1, W2, b2, w2h, out, E);
    } else {
        edge_net_wmma_kernel<false><<<blocks, THREADS, 0, stream>>>(
            h_w, ef, W1, b1, W2, b2, nullptr, out, E);
    }
}
